// Network_64063732187496
// MI455X (gfx1250) — compile-verified, matrix operand order checked by reading, not run on hardware
//
#include <hip/hip_runtime.h>
#include <hip/hip_bf16.h>

typedef __attribute__((ext_vector_type(16))) __bf16 v16bf;
typedef __attribute__((ext_vector_type(8)))  float  v8f;

#define NNODES 50000
#define NEDGES 800000

// ---------------------------------------------------------------- utilities

__device__ inline void atomAddF(float* p, float v) {
  __hip_atomic_fetch_add(p, v, __ATOMIC_RELAXED, __HIP_MEMORY_SCOPE_AGENT);
}

__global__ void zero_kernel(float* __restrict__ p, long n) {
  long i = (long)blockIdx.x * blockDim.x + threadIdx.x;
  long stride = (long)gridDim.x * blockDim.x;
  for (; i < n; i += stride) p[i] = 0.0f;
}

__global__ void deg_accum(const int* __restrict__ dst, float* __restrict__ deg, int E) {
  int i = blockIdx.x * blockDim.x + threadIdx.x;
  if (i < E) atomAddF(&deg[dst[i]], 1.0f);
}

__global__ void deg_fin(const float* __restrict__ deg, float* __restrict__ invd,
                        float* __restrict__ invs, int n) {
  int i = blockIdx.x * blockDim.x + threadIdx.x;
  if (i < n) {
    float d = fmaxf(deg[i], 1.0f);
    invd[i] = 1.0f / d;
    invs[i] = rsqrtf(d);
  }
}

__global__ void edge_norm_k(const int* __restrict__ src, const int* __restrict__ dst,
                            const float* __restrict__ invs, float* __restrict__ en, int E) {
  int i = blockIdx.x * blockDim.x + threadIdx.x;
  if (i < E) en[i] = invs[src[i]] * invs[dst[i]];
}

__global__ void softmax_k(const float* __restrict__ a, float* __restrict__ w) {
  int r = threadIdx.x;
  if (r >= 14) return;
  float mx = -1e30f;
  for (int j = 0; j < 8; ++j) mx = fmaxf(mx, a[r * 8 + j]);
  float e[8], s = 0.0f;
  for (int j = 0; j < 8; ++j) { e[j] = __expf(a[r * 8 + j] - mx); s += e[j]; }
  float inv = 1.0f / s;
  for (int j = 0; j < 8; ++j) w[r * 8 + j] = e[j] * inv;
}

// scatter: per (edge, 4-channel group): add msg to s_sum, msg*edge_norm to s_norm
__global__ void scatter_k(const float* __restrict__ h, const int* __restrict__ src,
                          const int* __restrict__ dst, const float* __restrict__ en,
                          float* __restrict__ ssum, float* __restrict__ snorm, int E) {
  long t = (long)blockIdx.x * blockDim.x + threadIdx.x;
  if (t >= (long)E * 16) return;
  int e = (int)(t >> 4), q = (int)(t & 15);
  int s = src[e], d = dst[e];
  float w = en[e];
  const float4 m = *(const float4*)(h + (long)s * 64 + q * 4);
  float* ps = ssum + (long)d * 64 + q * 4;
  float* pn = snorm + (long)d * 64 + q * 4;
  atomAddF(ps + 0, m.x); atomAddF(ps + 1, m.y);
  atomAddF(ps + 2, m.z); atomAddF(ps + 3, m.w);
  atomAddF(pn + 0, m.x * w); atomAddF(pn + 1, m.y * w);
  atomAddF(pn + 2, m.z * w); atomAddF(pn + 3, m.w * w);
}

// ---------------------------------------------------------------- BatchNorm

__global__ void bn_stats(const float* __restrict__ X, float* __restrict__ stats,
                         int Nv, int M) {
  int c = blockIdx.x;
  float s = 0.0f, s2 = 0.0f;
  for (int i = threadIdx.x; i < Nv; i += blockDim.x) {
    float v = X[(long)i * M + c];
    s += v; s2 += v * v;
  }
  __shared__ float rs[256], rq[256];
  rs[threadIdx.x] = s; rq[threadIdx.x] = s2;
  __syncthreads();
  for (int o = 128; o > 0; o >>= 1) {
    if (threadIdx.x < o) { rs[threadIdx.x] += rs[threadIdx.x + o]; rq[threadIdx.x] += rq[threadIdx.x + o]; }
    __syncthreads();
  }
  if (threadIdx.x == 0) {
    float m = rs[0] / (float)Nv;
    stats[c] = m;
    stats[256 + c] = rq[0] / (float)Nv - m * m;
  }
}

__global__ void bn_apply(float* __restrict__ X, const float* __restrict__ stats,
                         int Nv, int M, int doRelu) {
  long total = (long)Nv * M;
  long i = (long)blockIdx.x * blockDim.x + threadIdx.x;
  long stride = (long)gridDim.x * blockDim.x;
  for (; i < total; i += stride) {
    int c = (int)(i % M);
    float v = (X[i] - stats[c]) * rsqrtf(stats[256 + c] + 1e-5f);
    if (doRelu) v = fmaxf(v, 0.0f);
    X[i] = v;
  }
}

// ---------------------------------------------------------------- WMMA helpers

__device__ inline v8f v8zero() {
  v8f c;
#pragma unroll
  for (int i = 0; i < 8; ++i) c[i] = 0.0f;
  return c;
}

// Build A fragment (16x32 bf16) from row-major f32 matrix with leading dim ld.
// Lane l holds row (l&15); halves 0..7 = K k0+kb8+0..7, halves 8..15 = K k0+16+kb8+0..7
__device__ inline v16bf load_afrag(const float* __restrict__ X, int row0, int ld,
                                   int k0, int lane) {
  int r = row0 + (lane & 15);
  int kb8 = (lane >> 4) << 3;
  const float* p = X + (long)r * ld + k0 + kb8;
  v16bf a;
#pragma unroll
  for (int i = 0; i < 8; ++i) { a[i] = (__bf16)p[i]; a[8 + i] = (__bf16)p[16 + i]; }
  return a;
}

// swizzled LDS index for weight element (k within 32-block -> kr, column n)
__device__ inline int wswz(int kr, int n) {
  int lane = (((kr & 15) >> 3) << 4) | (n & 15);
  int h = (((kr >> 4) & 1) << 3) | (kr & 7);
  return lane * 16 + h;
}

// ---------------------------------------------------------------- generic GEMM
// C[Nv,M] = A[Nv,K] @ B[K,M]; bf16 WMMA, f32 accumulate. K%32==0, M%16==0, Nv%16==0.
template <int K, int M>
__global__ __launch_bounds__(128) void gemm_kernel(const float* __restrict__ A,
                                                   const float* __restrict__ B,
                                                   float* __restrict__ C, int Nv) {
  constexpr int KB = K / 32, NT = M / 16;
  __shared__ __bf16 wsh[KB * NT * 512];
  int tid = threadIdx.x;
  for (int idx = tid; idx < K * M; idx += 128) {
    int k = idx / M, n = idx % M;
    int kb = k >> 5, kr = k & 31, nt = n >> 4;
    wsh[((kb * NT + nt) << 9) + wswz(kr, n)] = (__bf16)B[idx];
  }
  __syncthreads();

  int tile = blockIdx.x * 4 + (tid >> 5);
  if (tile * 16 >= Nv) return;
  int lane = tid & 31;
  int row0 = tile * 16;

  v16bf af[KB];
#pragma unroll
  for (int kb = 0; kb < KB; ++kb) af[kb] = load_afrag(A, row0, K, kb * 32, lane);

  int coll = lane & 15;
  int m0 = row0 + ((lane >> 4) << 3);
#pragma unroll
  for (int nt = 0; nt < NT; ++nt) {
    v8f c = v8zero();
#pragma unroll
    for (int kb = 0; kb < KB; ++kb) {
      v16bf b = *(const v16bf*)(wsh + ((kb * NT + nt) << 9) + lane * 16);
      c = __builtin_amdgcn_wmma_f32_16x16x32_bf16(false, af[kb], false, b,
                                                  (short)0, c, false, false);
    }
#pragma unroll
    for (int i = 0; i < 8; ++i)
      C[(long)(m0 + i) * M + nt * 16 + coll] = c[i];
  }
}

// ---------------------------------------------------------------- fused mixed_op
// out += wv1*h + wv2*relu(sn@W0) + wv3*(h@W1 + sm@W2) + wv4*relu((h+ss)@W3)
//      + wv5*(h@W4 + ss@W5) + wv6*relu(h@W6) + wv7*relu((0.9*sn+0.1*x0)@W7)
__global__ __launch_bounds__(128) void mixed_kernel(
    const float* __restrict__ h, const float* __restrict__ ssum,
    const float* __restrict__ snorm, const float* __restrict__ x0,
    const float* __restrict__ invdeg,
    const float* __restrict__ w0, const float* __restrict__ w1,
    const float* __restrict__ w2, const float* __restrict__ w3,
    const float* __restrict__ w4, const float* __restrict__ w5,
    const float* __restrict__ w6, const float* __restrict__ w7,
    const float* __restrict__ wmix_row, float* __restrict__ out, int Nv) {
  __shared__ __bf16 wsh[8 * 2 * 4 * 512];  // 64 KB: 8 matrices, swizzled bf16
  int tid = threadIdx.x;
  const float* const Wsrc[8] = {w0, w1, w2, w3, w4, w5, w6, w7};
#pragma unroll
  for (int g = 0; g < 8; ++g) {
    const float* Wg = Wsrc[g];
    for (int idx = tid; idx < 4096; idx += 128) {
      int k = idx >> 6, n = idx & 63;
      int kb = k >> 5, kr = k & 31, nt = n >> 4;
      wsh[(((g * 2 + kb) * 4 + nt) << 9) + wswz(kr, n)] = (__bf16)Wg[idx];
    }
  }
  __syncthreads();

  int tile = blockIdx.x * 4 + (tid >> 5);
  if (tile * 16 >= Nv) return;
  int lane = tid & 31;
  int row0 = tile * 16;
  int r = row0 + (lane & 15);
  int kb8 = (lane >> 4) << 3;
  float invd = invdeg[r];

  // 12 A fragments: {h, s_norm, s_mean, s_sum, h+s_sum, 0.9*sn+0.1*x0} x 2 k-blocks
  v16bf fh[2], fnm[2], fme[2], fsu[2], fhs[2], fgc[2];
#pragma unroll
  for (int kb = 0; kb < 2; ++kb) {
    int k0 = kb * 32 + kb8;
    const float* ph = h + (long)r * 64 + k0;
    const float* ps = ssum + (long)r * 64 + k0;
    const float* pn = snorm + (long)r * 64 + k0;
    const float* px = x0 + (long)r * 64 + k0;
    float vh[16], vs[16], vn[16], vx[16];
#pragma unroll
    for (int i = 0; i < 8; ++i) {
      vh[i] = ph[i];  vh[8 + i] = ph[16 + i];
      vs[i] = ps[i];  vs[8 + i] = ps[16 + i];
      vn[i] = pn[i];  vn[8 + i] = pn[16 + i];
      vx[i] = px[i];  vx[8 + i] = px[16 + i];
    }
    v16bf a, b, c, d, e, f;
#pragma unroll
    for (int i = 0; i < 16; ++i) {
      a[i] = (__bf16)vh[i];
      b[i] = (__bf16)vn[i];
      c[i] = (__bf16)(vs[i] * invd);
      d[i] = (__bf16)vs[i];
      e[i] = (__bf16)(vh[i] + vs[i]);
      f[i] = (__bf16)(0.9f * vn[i] + 0.1f * vx[i]);
    }
    fh[kb] = a; fnm[kb] = b; fme[kb] = c; fsu[kb] = d; fhs[kb] = e; fgc[kb] = f;
  }

  float wv1 = wmix_row[1], wv2 = wmix_row[2], wv3 = wmix_row[3], wv4 = wmix_row[4];
  float wv5 = wmix_row[5], wv6 = wmix_row[6], wv7 = wmix_row[7];
  int coll = lane & 15;
  int m0 = row0 + ((lane >> 4) << 3);

#pragma unroll
  for (int nt = 0; nt < 4; ++nt) {
    int col = nt * 16 + coll;
    v8f acc;
#pragma unroll
    for (int i = 0; i < 8; ++i) acc[i] = wv1 * h[(long)(m0 + i) * 64 + col];

#define RUN_GEMM(FRAG, G, WV, RELU)                                              \
    {                                                                            \
      v8f c = v8zero();                                                          \
      _Pragma("unroll")                                                          \
      for (int kb = 0; kb < 2; ++kb) {                                           \
        v16bf b = *(const v16bf*)(wsh + ((((G) * 2 + kb) * 4 + nt) << 9) +       \
                                  lane * 16);                                    \
        c = __builtin_amdgcn_wmma_f32_16x16x32_bf16(false, FRAG[kb], false, b,   \
                                                    (short)0, c, false, false);  \
      }                                                                          \
      _Pragma("unroll")                                                          \
      for (int i = 0; i < 8; ++i) {                                              \
        float t = (RELU) ? fmaxf(c[i], 0.0f) : c[i];                             \
        acc[i] += (WV) * t;                                                      \
      }                                                                          \
    }

    RUN_GEMM(fnm, 0, wv2, 1)  // gcn
    RUN_GEMM(fh,  1, wv3, 0)  // sage self
    RUN_GEMM(fme, 2, wv3, 0)  // sage neighbor (mean)
    RUN_GEMM(fhs, 3, wv4, 1)  // gin
    RUN_GEMM(fh,  4, wv5, 0)  // gc1
    RUN_GEMM(fsu, 5, wv5, 0)  // gc2
    RUN_GEMM(fh,  6, wv6, 1)  // mlp
    RUN_GEMM(fgc, 7, wv7, 1)  // gcnii
#undef RUN_GEMM

#pragma unroll
    for (int i = 0; i < 8; ++i) {
      long o = (long)(m0 + i) * 64 + col;
      out[o] += acc[i];
    }
  }
}

// ---------------------------------------------------------------- concat + classifier

__global__ void copy_strided(const float* __restrict__ src, float* __restrict__ dst,
                             int n, int coff) {
  long i = (long)blockIdx.x * blockDim.x + threadIdx.x;
  if (i >= (long)n * 64) return;
  dst[(i >> 6) * 256 + coff + (i & 63)] = src[i];
}

__global__ void cls_k(const float* __restrict__ s1, const float* __restrict__ W,
                      const float* __restrict__ bias, float* __restrict__ out, int Nv) {
  int n = blockIdx.x;
  if (n >= Nv) return;
  int tid = threadIdx.x;
  __shared__ float red[64];
  __shared__ float pooled;
  float p = 0.0f;
  for (int k = tid; k < 256; k += 64) p += s1[(long)n * 256 + k];
  red[tid] = p;
  __syncthreads();
  for (int o = 32; o > 0; o >>= 1) {
    if (tid < o) red[tid] += red[tid + o];
    __syncthreads();
  }
  if (tid == 0) pooled = red[0] * (1.0f / 256.0f);
  __syncthreads();
  if (tid < 40) {
    float acc = bias[tid] + pooled * W[tid];
    for (int k = 0; k < 256; ++k) acc += s1[(long)n * 256 + k] * W[(1 + k) * 40 + tid];
    out[(long)n * 40 + tid] = acc;
  }
}

// ---------------------------------------------------------------- host orchestration

extern "C" void kernel_launch(void* const* d_in, const int* in_sizes, int n_in,
                              void* d_out, int out_size, void* d_ws, size_t ws_size,
                              hipStream_t stream) {
  (void)in_sizes; (void)n_in; (void)out_size; (void)ws_size;
  const int Nv = NNODES, E = NEDGES;

  const float* x       = (const float*)d_in[0];
  const int*   ei      = (const int*)d_in[1];
  const int*   src     = ei;
  const int*   dstp    = ei + E;
  const float* alphas  = (const float*)d_in[2];
  const float* stem_W  = (const float*)d_in[3];
  const float* pre_W   = (const float*)d_in[4];
  const float* pre0_W0 = (const float*)d_in[5];
  const float* pre1_W0 = (const float*)d_in[6];
  const float* pre0_W1 = (const float*)d_in[7];
  const float* pre1_W1 = (const float*)d_in[8];
  const float* Wb[8]   = {(const float*)d_in[9],  (const float*)d_in[10],
                          (const float*)d_in[11], (const float*)d_in[12],
                          (const float*)d_in[13], (const float*)d_in[14],
                          (const float*)d_in[15], (const float*)d_in[16]};
  const float* cls_W   = (const float*)d_in[17];
  const float* cls_b   = (const float*)d_in[18];

  float* ws = (float*)d_ws;
  size_t off = 0;
  auto alloc = [&](size_t n) { float* p = ws + off; off += (n + 255) & ~(size_t)255; return p; };
  float* deg   = alloc(Nv);
  float* invd  = alloc(Nv);
  float* invs  = alloc(Nv);
  float* en    = alloc(E);
  float* wmix  = alloc(112);
  float* stats = alloc(512);
  float* stem  = alloc((size_t)Nv * 192);
  float* x0    = alloc((size_t)Nv * 64);
  float* cell0 = alloc((size_t)Nv * 256);
  float* cell1 = alloc((size_t)Nv * 256);
  float* st[6];
  for (int i = 0; i < 6; ++i) st[i] = alloc((size_t)Nv * 64);
  float* ssum  = alloc((size_t)Nv * 64);
  float* snorm = alloc((size_t)Nv * 64);

  const int gblocks = (Nv / 16 + 3) / 4;  // 782 blocks of 4 waves (16-row tiles)

  // ---- degrees / edge norm / softmax ----
  zero_kernel<<<256, 256, 0, stream>>>(deg, Nv);
  deg_accum<<<(E + 255) / 256, 256, 0, stream>>>(dstp, deg, E);
  deg_fin<<<(Nv + 255) / 256, 256, 0, stream>>>(deg, invd, invs, Nv);
  edge_norm_k<<<(E + 255) / 256, 256, 0, stream>>>(src, dstp, invs, en, E);
  softmax_k<<<1, 32, 0, stream>>>(alphas, wmix);

  // ---- stem & preprocess ----
  gemm_kernel<128, 192><<<gblocks, 128, 0, stream>>>(x, stem_W, stem, Nv);
  bn_stats<<<192, 256, 0, stream>>>(stem, stats, Nv, 192);
  bn_apply<<<2048, 256, 0, stream>>>(stem, stats, Nv, 192, 0);

  gemm_kernel<128, 64><<<gblocks, 128, 0, stream>>>(x, pre_W, x0, Nv);
  bn_stats<<<64, 256, 0, stream>>>(x0, stats, Nv, 64);
  bn_apply<<<2048, 256, 0, stream>>>(x0, stats, Nv, 64, 1);

  const float* s0 = stem;
  const float* s1 = stem;
  float* couts[2] = {cell0, cell1};

  for (int ci = 0; ci < 2; ++ci) {
    // a = relu(bn(s0 @ p0)), b = relu(bn(s1 @ p1))
    if (ci == 0) gemm_kernel<192, 64><<<gblocks, 128, 0, stream>>>(s0, pre0_W0, st[0], Nv);
    else         gemm_kernel<192, 64><<<gblocks, 128, 0, stream>>>(s0, pre0_W1, st[0], Nv);
    bn_stats<<<64, 256, 0, stream>>>(st[0], stats, Nv, 64);
    bn_apply<<<2048, 256, 0, stream>>>(st[0], stats, Nv, 64, 1);

    if (ci == 0) gemm_kernel<192, 64><<<gblocks, 128, 0, stream>>>(s1, pre1_W0, st[1], Nv);
    else         gemm_kernel<256, 64><<<gblocks, 128, 0, stream>>>(s1, pre1_W1, st[1], Nv);
    bn_stats<<<64, 256, 0, stream>>>(st[1], stats, Nv, 64);
    bn_apply<<<2048, 256, 0, stream>>>(st[1], stats, Nv, 64, 1);

    int offset = 0;
    for (int step = 0; step < 4; ++step) {
      int ns = 2 + step;
      float* outp = st[2 + step];
      zero_kernel<<<2048, 256, 0, stream>>>(outp, (long)Nv * 64);
      for (int j = 0; j < ns; ++j) {
        const float* h = st[j];
        zero_kernel<<<2048, 256, 0, stream>>>(ssum, (long)Nv * 64);
        zero_kernel<<<2048, 256, 0, stream>>>(snorm, (long)Nv * 64);
        scatter_k<<<(int)(((long)E * 16 + 255) / 256), 256, 0, stream>>>(
            h, src, dstp, en, ssum, snorm, E);
        int slot = offset + j;
        size_t woff = (size_t)(ci * 14 + slot) * 4096;
        mixed_kernel<<<gblocks, 128, 0, stream>>>(
            h, ssum, snorm, x0, invd,
            Wb[0] + woff, Wb[1] + woff, Wb[2] + woff, Wb[3] + woff,
            Wb[4] + woff, Wb[5] + woff, Wb[6] + woff, Wb[7] + woff,
            wmix + slot * 8, outp, Nv);
      }
      offset += ns;
    }
    // out_cell = concat(states[-4:]) -> [N,256]
    for (int q = 0; q < 4; ++q)
      copy_strided<<<(Nv * 64 + 255) / 256, 256, 0, stream>>>(st[2 + q], couts[ci], Nv, q * 64);
    s0 = s1;
    s1 = couts[ci];
  }

  cls_k<<<Nv, 64, 0, stream>>>(s1, cls_W, cls_b, (float*)d_out, Nv);
}
